// Decoder_22273700397282
// MI455X (gfx1250) — matrix operand order repaired and checked by reading, not compiled, not run on hardware
//
#include <hip/hip_runtime.h>

// ---------------------------------------------------------------------------
// Problem constants (from reference)
// ---------------------------------------------------------------------------
#define NBATCH 32
#define SEQ    512
#define DIN    1024
#define DH     2048   // 2*IN
#define DA     3072   // 3*IN = OUT_SIZE

// GEMM tiling
#define KC      512                 // K chunk staged in LDS
#define APAD    8                   // bf16 pad elements per row
#define APITCH  (KC + APAD)         // 520
#define EPITCH  (DA + 8)            // 3080
#define MT      32                  // rows per workgroup
#define GEMM_THREADS 512            // 16 waves

// LDS carve (bytes)
#define SM_AHI_OFF   0
#define SM_ALO_OFF   (MT * APITCH * 2)                       // 33280
#define SM_E_OFF     (2 * MT * APITCH * 2)                   // 66560
#define SM_ROWP_OFF  (SM_E_OFF + MT * EPITCH * 2)            // 263680
#define SM_INV_OFF   (SM_ROWP_OFF + MT * 8 * 4)              // 264704
#define SM_TOTAL     (SM_INV_OFF + MT * 4)                   // 264832

// ---------------------------------------------------------------------------
// Types for WMMA
// ---------------------------------------------------------------------------
typedef __attribute__((ext_vector_type(16))) __bf16        v16bf;
typedef __attribute__((ext_vector_type(8)))  float         v8f;
typedef __attribute__((ext_vector_type(8)))  unsigned int  v8u;
typedef __attribute__((ext_vector_type(4)))  unsigned int  v4u;

union FragA { v4u h[2]; v16bf b; };   // 32 bytes
union FragB { v8u w;    v16bf b; };   // 32 bytes

// RNE float -> bf16 split: f ~= hi + lo (each bf16)
__device__ __forceinline__ void split_bf16(float f, unsigned short& hi, unsigned short& lo) {
    unsigned u = __float_as_uint(f);
    unsigned r = u + 0x7fffu + ((u >> 16) & 1u);
    hi = (unsigned short)(r >> 16);
    float fh = __uint_as_float(((unsigned)hi) << 16);
    float d  = f - fh;                       // exact in fp32
    unsigned u2 = __float_as_uint(d);
    unsigned r2 = u2 + 0x7fffu + ((u2 >> 16) & 1u);
    lo = (unsigned short)(r2 >> 16);
}

// ---------------------------------------------------------------------------
// Kernel 1: transpose + split Wa (K=3072 x N=3072, row-major) into
//           WaT_hi / WaT_lo laid out (N, K) bf16 for contiguous B-frag loads.
// ---------------------------------------------------------------------------
__global__ void k_wat(const float* __restrict__ Wa,
                      unsigned short* __restrict__ wt_hi,
                      unsigned short* __restrict__ wt_lo) {
    __shared__ float tile[32][33];
    const int n0 = blockIdx.x * 32, k0 = blockIdx.y * 32;
    const int tx = threadIdx.x, ty = threadIdx.y;
    #pragma unroll
    for (int j = ty; j < 32; j += 8)
        tile[j][tx] = Wa[(size_t)(k0 + j) * DA + (n0 + tx)];
    __syncthreads();
    #pragma unroll
    for (int j = ty; j < 32; j += 8) {
        float f = tile[tx][j];                 // = Wa[k0+tx][n0+j]
        unsigned short hi, lo;
        split_bf16(f, hi, lo);
        size_t o = (size_t)(n0 + j) * DA + (k0 + tx);
        wt_hi[o] = hi;
        wt_lo[o] = lo;
    }
}

// ---------------------------------------------------------------------------
// Kernel 2: s[b,k] = sum_s h_state[b,s,k]
// ---------------------------------------------------------------------------
__global__ void k_ssum(const float* __restrict__ h, float* __restrict__ sbuf) {
    const int b = blockIdx.y;
    const int k = blockIdx.x * 256 + threadIdx.x;
    const float* p = h + (size_t)b * SEQ * DH + k;
    float a = 0.f;
    for (int i = 0; i < SEQ; ++i) a += p[(size_t)i * DH];
    sbuf[b * DH + k] = a;
}

// ---------------------------------------------------------------------------
// Kernel 3: r[b,f] = ba[f] + bs[f] + sum_k s[b,k] * Ws[k,f]
// ---------------------------------------------------------------------------
__global__ void k_r(const float* __restrict__ sbuf, const float* __restrict__ Ws,
                    const float* __restrict__ ba, const float* __restrict__ bs,
                    float* __restrict__ rbuf) {
    const int b = blockIdx.y;
    const int f = blockIdx.x * 256 + threadIdx.x;
    float a = ba[f] + bs[f];
    const float* sp = sbuf + b * DH;
    const float* wp = Ws + f;
    for (int k = 0; k < DH; ++k) a += sp[k] * wp[(size_t)k * DA];
    rbuf[b * DA + f] = a;
}

// ---------------------------------------------------------------------------
// Kernel 4: per-batch compaction of mask==1 rows (deterministic order)
// ---------------------------------------------------------------------------
__global__ void k_compact(const int* __restrict__ mask,
                          int* __restrict__ rows, int* __restrict__ cnt1) {
    const int b = blockIdx.x;
    const int s = threadIdx.x;       // 512 threads = 16 waves
    const int pred = (mask[b * SEQ + s] != 0) ? 1 : 0;
    const unsigned lane = s & 31;
    const int w = s >> 5;
    unsigned m32 = (unsigned)__ballot(pred);
    __shared__ int wsum[16];
    __shared__ int wpre[16];
    if (lane == 0) wsum[w] = __popc(m32);
    __syncthreads();
    if (s == 0) {
        int run = 0;
        for (int i = 0; i < 16; ++i) { wpre[i] = run; run += wsum[i]; }
        cnt1[b] = run;
    }
    __syncthreads();
    if (pred) {
        int pos = __popc(m32 & ((1u << lane) - 1u));
        rows[b * SEQ + wpre[w] + pos] = s;
    }
}

// ---------------------------------------------------------------------------
// Kernel 5: fused GEMM (bf16x3 WMMA) + tanh + exp + row-softmax + column sum.
// Block (rb, b): 32 compacted rows of batch b, full N=3072, K=3072.
// 16 waves: wm = wave>>3 (M half of 16 rows), wn = wave&7 (384-col slab,
// processed in 3 passes of 128 cols = 8 WMMA tiles).
// Writes partial[b, rb, f] = sum_{rows in block} softmax_f(tanh(et))[f].
// ---------------------------------------------------------------------------
__global__ void __launch_bounds__(GEMM_THREADS)
k_gemm(const float* __restrict__ h, const float* __restrict__ x,
       const unsigned short* __restrict__ wt_hi,
       const unsigned short* __restrict__ wt_lo,
       const float* __restrict__ rbuf, const int* __restrict__ rows,
       const int* __restrict__ cnt1, float* __restrict__ partial) {
    const int b  = blockIdx.y;
    const int rb = blockIdx.x;
    const int c1 = cnt1[b];
    if (rb * MT >= c1) return;    // uniform early exit

    extern __shared__ char smem[];
    unsigned short* smA_hi = (unsigned short*)(smem + SM_AHI_OFF);
    unsigned short* smA_lo = (unsigned short*)(smem + SM_ALO_OFF);
    _Float16*       smE    = (_Float16*)     (smem + SM_E_OFF);
    float*          smRowP = (float*)        (smem + SM_ROWP_OFF);  // [32][8]
    float*          smInv  = (float*)        (smem + SM_INV_OFF);   // [32]

    const int tid   = threadIdx.x;
    const int lane  = tid & 31;
    const int w     = tid >> 5;
    const int wm    = w >> 3;        // 0..1
    const int wn    = w & 7;         // 0..7
    const int l15   = lane & 15;
    const int khalf = lane >> 4;

    if (tid < MT * 8) smRowP[tid] = 0.0f;

    // cooperative A staging assignment: row + 32-col segment per thread
    const int ar   = tid >> 4;                 // 0..31
    const int aseg = (tid & 15) * 32;          // 0..480
    const int gi   = rb * MT + ar;
    const int gic  = (gi < c1) ? gi : (c1 - 1);   // clamp (excluded later)
    const int sidx = rows[b * SEQ + gic];
    const float* hrow = h + ((size_t)(b * SEQ + sidx)) * DH;
    const float* xrow = x + ((size_t)(b * SEQ + sidx)) * DIN;

    const int mloc = wm * 16 + l15;            // A-frag row in LDS
    const v8f vzero = {0.f,0.f,0.f,0.f,0.f,0.f,0.f,0.f};

    for (int p = 0; p < 3; ++p) {
        v8f acc[8];
        #pragma unroll
        for (int t = 0; t < 8; ++t) acc[t] = vzero;

        const int n0 = wn * 384 + p * 128;

        for (int kc = 0; kc < 6; ++kc) {
            __syncthreads();
            // ---- stage + split A chunk (32 rows x 512 k) into LDS ----
            {
                const int cbase = kc * KC + aseg;
                const float* src = (cbase < DH) ? (hrow + cbase)
                                                : (xrow + (cbase - DH));
                unsigned short* dh = smA_hi + ar * APITCH + aseg;
                unsigned short* dl = smA_lo + ar * APITCH + aseg;
                #pragma unroll
                for (int i = 0; i < 32; i += 4) {
                    float4 f4 = *(const float4*)(src + i);
                    unsigned short h0,h1,h2,h3,l0,l1,l2,l3;
                    split_bf16(f4.x, h0, l0);
                    split_bf16(f4.y, h1, l1);
                    split_bf16(f4.z, h2, l2);
                    split_bf16(f4.w, h3, l3);
                    *(unsigned int*)(dh + i)     = (unsigned)h0 | ((unsigned)h1 << 16);
                    *(unsigned int*)(dh + i + 2) = (unsigned)h2 | ((unsigned)h3 << 16);
                    *(unsigned int*)(dl + i)     = (unsigned)l0 | ((unsigned)l1 << 16);
                    *(unsigned int*)(dl + i + 2) = (unsigned)l2 | ((unsigned)l3 << 16);
                }
            }
            __syncthreads();

            // ---- K inner loop: 16 steps of K=32 ----
            for (int kt = 0; kt < 16; ++kt) {
                // A frags from LDS (contiguous K order matching B: per lane
                // 16 consecutive K at khalf*16)
                const unsigned short* pah = smA_hi + mloc * APITCH + kt * 32 + khalf * 16;
                const unsigned short* pal = smA_lo + mloc * APITCH + kt * 32 + khalf * 16;
                FragA ah, al;
                ah.h[0] = *(const v4u*)(pah);
                ah.h[1] = *(const v4u*)(pah + 8);
                al.h[0] = *(const v4u*)(pal);
                al.h[1] = *(const v4u*)(pal + 8);

                const size_t kk = (size_t)kc * KC + (size_t)kt * 32 + (size_t)khalf * 16;
                #pragma unroll
                for (int t = 0; t < 8; ++t) {
                    const size_t nn = (size_t)(n0 + t * 16 + l15);
                    FragB bh, bl;
                    bh.w = *(const v8u*)(wt_hi + nn * DA + kk);
                    bl.w = *(const v8u*)(wt_lo + nn * DA + kk);
                    acc[t] = __builtin_amdgcn_wmma_f32_16x16x32_bf16(
                        false, ah.b, false, bh.b, (short)0, acc[t], false, false);
                    acc[t] = __builtin_amdgcn_wmma_f32_16x16x32_bf16(
                        false, al.b, false, bh.b, (short)0, acc[t], false, false);
                    acc[t] = __builtin_amdgcn_wmma_f32_16x16x32_bf16(
                        false, ah.b, false, bl.b, (short)0, acc[t], false, false);
                }
            }
        }

        // ---- epilogue for this pass: e = exp(tanh(et)), stash in LDS,
        //      accumulate per-row sums deterministically per wave slot ----
        #pragma unroll
        for (int t = 0; t < 8; ++t) {
            const int n = n0 + t * 16 + l15;
            const float rv = rbuf[b * DA + n];
            #pragma unroll
            for (int v = 0; v < 8; ++v) {
                const int mrow = wm * 16 + khalf * 8 + v;   // C layout: M=(lane>>4)*8+v
                float e = expf(tanhf(acc[t][v] + rv));
                smE[mrow * EPITCH + n] = (_Float16)e;
                float red = e;
                red += __shfl_xor(red, 1);
                red += __shfl_xor(red, 2);
                red += __shfl_xor(red, 4);
                red += __shfl_xor(red, 8);
                if (l15 == 0) smRowP[mrow * 8 + wn] += red;  // wave-private slot
            }
        }
    }

    __syncthreads();
    if (tid < MT) {
        float rs = 0.f;
        #pragma unroll
        for (int j = 0; j < 8; ++j) rs += smRowP[tid * 8 + j];  // fixed order
        const int valid = (rb * MT + tid) < c1;
        smInv[tid] = valid ? (1.0f / rs) : 0.0f;
    }
    __syncthreads();

    // ---- column sums of softmax rows -> partial[b, rb, :] ----
    float* outp = partial + ((size_t)b * 16 + rb) * DA;
    for (int f = tid; f < DA; f += GEMM_THREADS) {
        float a = 0.f;
        #pragma unroll
        for (int m = 0; m < MT; ++m)
            a += (float)smE[m * EPITCH + f] * smInv[m];
        outp[f] = a;
    }
}

// ---------------------------------------------------------------------------
// Kernel 6: finalize: out[b,f] = trigger_a[b,f] *
//           (cnt0/3072 + sum_{rb<nb} partial[b,rb,f])
// ---------------------------------------------------------------------------
__global__ void k_final(const float* __restrict__ h, const float* __restrict__ x,
                        const int* __restrict__ trigger, const int* __restrict__ cnt1,
                        const float* __restrict__ partial, float* __restrict__ out) {
    const int b = blockIdx.y;
    const int f = blockIdx.x * 256 + threadIdx.x;
    const int c1 = cnt1[b];
    const int nb = (c1 + MT - 1) / MT;
    float a = (float)(SEQ - c1) * (1.0f / (float)DA);
    const float* pp = partial + (size_t)b * 16 * DA + f;
    for (int rb = 0; rb < nb; ++rb) a += pp[(size_t)rb * DA];
    const int st = trigger[b];
    float ta = (f < DH) ? h[((size_t)b * SEQ + st) * DH + f]
                        : x[((size_t)b * SEQ + st) * DIN + (f - DH)];
    out[b * DA + f] = ta * a;
}

// ---------------------------------------------------------------------------
// Launch
// ---------------------------------------------------------------------------
extern "C" void kernel_launch(void* const* d_in, const int* in_sizes, int n_in,
                              void* d_out, int out_size, void* d_ws, size_t ws_size,
                              hipStream_t stream) {
    const float* h    = (const float*)d_in[0];
    const float* x    = (const float*)d_in[1];
    const int*   trig = (const int*)  d_in[2];
    const int*   mask = (const int*)  d_in[3];
    const float* Wa   = (const float*)d_in[4];
    const float* ba   = (const float*)d_in[5];
    const float* Ws   = (const float*)d_in[6];
    const float* bs   = (const float*)d_in[7];
    float* out = (float*)d_out;

    char* ws = (char*)d_ws;
    size_t off = 0;
    auto carve = [&](size_t bytes) {
        char* p = ws + off;
        off += (bytes + 255) & ~(size_t)255;
        return p;
    };
    unsigned short* wt_hi  = (unsigned short*)carve((size_t)DA * DA * 2);
    unsigned short* wt_lo  = (unsigned short*)carve((size_t)DA * DA * 2);
    float*          sbuf   = (float*)carve((size_t)NBATCH * DH * 4);
    float*          rbuf   = (float*)carve((size_t)NBATCH * DA * 4);
    int*            rows   = (int*)  carve((size_t)NBATCH * SEQ * 4);
    int*            cnt1   = (int*)  carve((size_t)NBATCH * 4);
    float*          part   = (float*)carve((size_t)NBATCH * 16 * DA * 4);
    (void)ws_size; (void)in_sizes; (void)n_in; (void)out_size;

    k_wat    <<<dim3(DA / 32, DA / 32), dim3(32, 8), 0, stream>>>(Wa, wt_hi, wt_lo);
    k_ssum   <<<dim3(DH / 256, NBATCH), 256, 0, stream>>>(h, sbuf);
    k_compact<<<NBATCH, SEQ, 0, stream>>>(mask, rows, cnt1);
    k_r      <<<dim3(DA / 256, NBATCH), 256, 0, stream>>>(sbuf, Ws, ba, bs, rbuf);
    k_gemm   <<<dim3(SEQ / MT, NBATCH), GEMM_THREADS, SM_TOTAL, stream>>>(
                 h, x, wt_hi, wt_lo, rbuf, rows, cnt1, part);
    k_final  <<<dim3(DA / 256, NBATCH), 256, 0, stream>>>(h, x, trig, cnt1, part, out);
}
